// WhisperAttentionTVM_2456721293849
// MI455X (gfx1250) — matrix operand order repaired and checked by reading, not compiled, not run on hardware
//
#include <hip/hip_runtime.h>

typedef __bf16 bf16;
typedef __attribute__((ext_vector_type(8)))  __bf16 bf16x8;
typedef __attribute__((ext_vector_type(16))) __bf16 bf16x16;
typedef __attribute__((ext_vector_type(8)))  float  f32x8;

namespace {
constexpr int kB  = 4;
constexpr int kS  = 1500;
constexpr int kE  = 1024;
constexpr int kH  = 16;
constexpr int kD  = 64;
constexpr int kSP = 1504;          // padded key-stride for V^T rows (32B-aligned b128 loads)
constexpr int kM  = kB * kS;       // 6000 rows
constexpr int kMT = 188;           // ceil(6000/32) row tiles of 32
constexpr float kScale = 0.125f;   // HEAD_DIM^-0.5
}

__device__ __forceinline__ f32x8 wmma_bf16(bf16x16 a, bf16x16 b, f32x8 c) {
  return __builtin_amdgcn_wmma_f32_16x16x32_bf16(false, a, false, b, (short)0, c, false, false);
}

__device__ __forceinline__ bf16x16 cat8(bf16x8 lo, bf16x8 hi) {
  return __builtin_shufflevector(lo, hi, 0,1,2,3,4,5,6,7,8,9,10,11,12,13,14,15);
}

// fp32 -> bf16, 8 elements/thread (b128 in / b128 out)
__global__ void cvt_f32_to_bf16_x8(const float* __restrict__ src, bf16* __restrict__ dst, int n8) {
  int i = blockIdx.x * blockDim.x + threadIdx.x;
  if (i < n8) {
    f32x8 v = *(const f32x8*)(src + (size_t)i * 8);
    bf16x8 o;
#pragma unroll
    for (int j = 0; j < 8; ++j) o[j] = (bf16)v[j];
    *(bf16x8*)(dst + (size_t)i * 8) = o;
  }
}

// y[m,n] = sum_k X[m,k] * W[n,k] (+bias)(*scale); each wave computes a 32x64 tile,
// software-pipelined (next k-step's A/B loads issued before current WMMAs).
// mode 0: Q  -> bf16 [B,H,S,D], (acc+b)*scale
// mode 1: K  -> bf16 [B,H,S,D], acc
// mode 2: V  -> bf16 [B,H,D,SP] (transposed, padded), acc+b
// mode 3: out-> f32  [M,E], acc+b
__global__ __launch_bounds__(256) void gemm_bf16_32x64(
    const bf16* __restrict__ X, const bf16* __restrict__ W,
    const float* __restrict__ bias, void* __restrict__ outp,
    float scale, int mode) {
  const int wid  = blockIdx.x * 8 + (threadIdx.x >> 5);
  const int lane = threadIdx.x & 31;
  const int half = lane >> 4;
  const int ln   = lane & 15;
  const int nchunk = wid & 15;   // 16 chunks of 64 columns
  const int mtile  = wid >> 4;   // 188 row tiles of 32
  if (mtile >= kMT) return;      // wave-uniform

  const int m0 = mtile * 32;
  const int r0 = m0 + ln;
  const int r1 = m0 + 16 + ln;
  const bf16* __restrict__ arow0 = X + (size_t)(r0 < kM ? r0 : kM - 1) * kE;
  const bf16* __restrict__ arow1 = X + (size_t)(r1 < kM ? r1 : kM - 1) * kE;
  const int n0 = nchunk * 64;
  const bf16* __restrict__ wbase = W + (size_t)(n0 + ln) * kE + 16 * half;

  f32x8 acc[8] = {};   // [mt*4 + nt]

  auto ld_a = [&](const bf16* arow, int kk) -> bf16x16 {
    return cat8(*(const bf16x8*)(arow + kk + 16 * half),
                *(const bf16x8*)(arow + kk + 16 * half + 8));
  };

  // prologue: k-step 0 in flight
  bf16x16 a0 = ld_a(arow0, 0);
  bf16x16 a1 = ld_a(arow1, 0);
  bf16x16 b0 = *(const bf16x16*)(wbase + (size_t)0 * 16 * kE);
  bf16x16 b1 = *(const bf16x16*)(wbase + (size_t)1 * 16 * kE);
  bf16x16 b2 = *(const bf16x16*)(wbase + (size_t)2 * 16 * kE);
  bf16x16 b3 = *(const bf16x16*)(wbase + (size_t)3 * 16 * kE);

#pragma unroll 4
  for (int kk = 0; kk < kE; kk += 32) {
    const int kn = (kk + 32 < kE) ? kk + 32 : 0;   // wrap-safe dummy prefetch on last iter
    // issue next k-step's loads first so they overlap the WMMAs below
    bf16x16 na0 = ld_a(arow0, kn);
    bf16x16 na1 = ld_a(arow1, kn);
    bf16x16 nb0 = *(const bf16x16*)(wbase + kn + (size_t)0 * 16 * kE);
    bf16x16 nb1 = *(const bf16x16*)(wbase + kn + (size_t)1 * 16 * kE);
    bf16x16 nb2 = *(const bf16x16*)(wbase + kn + (size_t)2 * 16 * kE);
    bf16x16 nb3 = *(const bf16x16*)(wbase + kn + (size_t)3 * 16 * kE);

    acc[0] = wmma_bf16(a0, b0, acc[0]);
    acc[1] = wmma_bf16(a0, b1, acc[1]);
    acc[2] = wmma_bf16(a0, b2, acc[2]);
    acc[3] = wmma_bf16(a0, b3, acc[3]);
    acc[4] = wmma_bf16(a1, b0, acc[4]);
    acc[5] = wmma_bf16(a1, b1, acc[5]);
    acc[6] = wmma_bf16(a1, b2, acc[6]);
    acc[7] = wmma_bf16(a1, b3, acc[7]);

    a0 = na0; a1 = na1;
    b0 = nb0; b1 = nb1; b2 = nb2; b3 = nb3;
  }

#pragma unroll
  for (int mt = 0; mt < 2; ++mt) {
#pragma unroll
    for (int nt = 0; nt < 4; ++nt) {
      const int n = n0 + nt * 16 + ln;
      const float bv = bias ? bias[n] : 0.0f;
#pragma unroll
      for (int r = 0; r < 8; ++r) {
        const int mr = m0 + mt * 16 + r + 8 * half;
        if (mr >= kM) continue;
        const float v = acc[mt * 4 + nt][r];
        if (mode == 3) {
          ((float*)outp)[(size_t)mr * kE + n] = v + bv;
        } else {
          const int bb = mr / kS;
          const int ss = mr - bb * kS;
          const int hh = n >> 6;
          const int dd = n & 63;
          float o;
          if (mode == 0)      o = (v + bv) * scale;
          else if (mode == 1) o = v;
          else                o = v + bv;
          size_t idx;
          if (mode == 2) idx = (((size_t)bb * kH + hh) * kD + dd) * kSP + ss;
          else           idx = (((size_t)bb * kH + hh) * kS + ss) * kD + dd;
          ((bf16*)outp)[idx] = (bf16)o;
        }
      }
    }
  }
}

// Flash attention: one wave per 16-row Q tile, streaming keys in chunks of 32.
// K loads for chunk k0+32 and V loads for chunk k0 are issued right after the
// score WMMAs, so they fly under the softmax VALU work.
__global__ __launch_bounds__(256) void flash_attn(
    const bf16* __restrict__ Q, const bf16* __restrict__ Kb,
    const bf16* __restrict__ Vt, const float* __restrict__ mask,
    bf16* __restrict__ attn_out) {
  __shared__ bf16 lds_p[8][16 * 32];   // per-wave 16x32 prob tile (C-layout -> A-layout)
  const int wave = threadIdx.x >> 5;
  const int lane = threadIdx.x & 31;
  const int half = lane >> 4;
  const int ln   = lane & 15;
  const int bh   = blockIdx.x / 12;
  const int qblk = blockIdx.x % 12;
  const int b = bh >> 4;
  const int h = bh & 15;
  const int q0 = qblk * 128 + wave * 16;

  const bf16* __restrict__ Qbase = Q  + (size_t)bh * kS * kD;
  const bf16* __restrict__ Kbase = Kb + (size_t)bh * kS * kD;
  const bf16* __restrict__ Vbase = Vt + (size_t)bh * kD * kSP;

  int qm = q0 + ln; if (qm > kS - 1) qm = kS - 1;
  const bf16* qrow = Qbase + (size_t)qm * kD;
  const bf16x16 qa0 = cat8(*(const bf16x8*)(qrow +  0 + 16 * half),
                           *(const bf16x8*)(qrow +  8 + 16 * half));   // d 0..31
  const bf16x16 qa1 = cat8(*(const bf16x8*)(qrow + 32 + 16 * half),
                           *(const bf16x8*)(qrow + 40 + 16 * half));   // d 32..63

  f32x8 o[4] = {};
  float mrow[8], lrow[8];
#pragma unroll
  for (int r = 0; r < 8; ++r) { mrow[r] = -3.0e38f; lrow[r] = 0.0f; }
  int qclamp[8];
#pragma unroll
  for (int r = 0; r < 8; ++r) {
    int q = q0 + 8 * half + r;
    qclamp[r] = q > kS - 1 ? kS - 1 : q;
  }

  auto kclamp = [&](int key) { return key > kS - 1 ? kS - 1 : key; };

  // prologue: K tiles for chunk 0
  const bf16* kra = Kbase + (size_t)kclamp(ln) * kD + 16 * half;
  const bf16* krb = Kbase + (size_t)kclamp(16 + ln) * kD + 16 * half;
  bf16x16 b0lo = *(const bf16x16*)(kra);
  bf16x16 b0hi = *(const bf16x16*)(kra + 32);
  bf16x16 b1lo = *(const bf16x16*)(krb);
  bf16x16 b1hi = *(const bf16x16*)(krb + 32);

  for (int k0 = 0; k0 < kS; k0 += 32) {
    f32x8 s0 = {}, s1 = {};
    s0 = wmma_bf16(qa0, b0lo, s0);
    s0 = wmma_bf16(qa1, b0hi, s0);
    s1 = wmma_bf16(qa0, b1lo, s1);
    s1 = wmma_bf16(qa1, b1hi, s1);

    // prefetch next chunk's K tiles (clamped dummy loads past the end)
    {
      const bf16* nka = Kbase + (size_t)kclamp(k0 + 32 + ln) * kD + 16 * half;
      const bf16* nkb = Kbase + (size_t)kclamp(k0 + 48 + ln) * kD + 16 * half;
      b0lo = *(const bf16x16*)(nka);
      b0hi = *(const bf16x16*)(nka + 32);
      b1lo = *(const bf16x16*)(nkb);
      b1hi = *(const bf16x16*)(nkb + 32);
    }
    // prefetch current chunk's V tiles (consumed after softmax)
    bf16x16 vb0 = *(const bf16x16*)(Vbase + (size_t)( 0 + ln) * kSP + k0 + 16 * half);
    bf16x16 vb1 = *(const bf16x16*)(Vbase + (size_t)(16 + ln) * kSP + k0 + 16 * half);
    bf16x16 vb2 = *(const bf16x16*)(Vbase + (size_t)(32 + ln) * kSP + k0 + 16 * half);
    bf16x16 vb3 = *(const bf16x16*)(Vbase + (size_t)(48 + ln) * kSP + k0 + 16 * half);

    const float NEG = -3.0e38f;
    const int keyA = k0 + ln;
    const int keyB = k0 + 16 + ln;
    const bool vA = keyA < kS;
    const bool vB = keyB < kS;
    const int kac = kclamp(keyA);
    const int kbc = kclamp(keyB);
#pragma unroll
    for (int r = 0; r < 8; ++r) {
      const float* mr_ptr = mask + ((size_t)b * kS + qclamp[r]) * kS;
      const float ma = mr_ptr[kac];
      const float mb = mr_ptr[kbc];
      s0[r] = vA ? s0[r] + ma : NEG;
      s1[r] = vB ? s1[r] + mb : NEG;
    }

    // online softmax: row reductions within the 16-lane column groups
#pragma unroll
    for (int r = 0; r < 8; ++r) {
      float red = fmaxf(s0[r], s1[r]);
      red = fmaxf(red, __shfl_xor(red, 1));
      red = fmaxf(red, __shfl_xor(red, 2));
      red = fmaxf(red, __shfl_xor(red, 4));
      red = fmaxf(red, __shfl_xor(red, 8));
      const float mn   = fmaxf(mrow[r], red);
      const float corr = __expf(mrow[r] - mn);
      const float p0 = __expf(s0[r] - mn);
      const float p1 = __expf(s1[r] - mn);
      mrow[r] = mn;
      s0[r] = p0; s1[r] = p1;
      float t = p0 + p1;
      t += __shfl_xor(t, 1);
      t += __shfl_xor(t, 2);
      t += __shfl_xor(t, 4);
      t += __shfl_xor(t, 8);
      lrow[r] = lrow[r] * corr + t;
      o[0][r] *= corr; o[1][r] *= corr; o[2][r] *= corr; o[3][r] *= corr;
    }

    // C-layout probs -> LDS -> A-layout (same-wave DS ops are in-order)
    bf16* pl = &lds_p[wave][0];
#pragma unroll
    for (int r = 0; r < 8; ++r) {
      const int mrr = r + 8 * half;
      pl[mrr * 32 + ln]      = (bf16)s0[r];
      pl[mrr * 32 + 16 + ln] = (bf16)s1[r];
    }
    const bf16x16 pa = cat8(*(const bf16x8*)(pl + ln * 32 + 16 * half),
                            *(const bf16x8*)(pl + ln * 32 + 16 * half + 8));

    o[0] = wmma_bf16(pa, vb0, o[0]);
    o[1] = wmma_bf16(pa, vb1, o[1]);
    o[2] = wmma_bf16(pa, vb2, o[2]);
    o[3] = wmma_bf16(pa, vb3, o[3]);
  }

  float inv[8];
#pragma unroll
  for (int r = 0; r < 8; ++r) inv[r] = 1.0f / lrow[r];
#pragma unroll
  for (int dt = 0; dt < 4; ++dt) {
#pragma unroll
    for (int r = 0; r < 8; ++r) {
      const int q = q0 + r + 8 * half;
      if (q < kS) {
        const int d = dt * 16 + ln;
        attn_out[(((size_t)b * kS + q) * kH + h) * kD + d] = (bf16)(o[dt][r] * inv[r]);
      }
    }
  }
}

extern "C" void kernel_launch(void* const* d_in, const int* in_sizes, int n_in,
                              void* d_out, int out_size, void* d_ws, size_t ws_size,
                              hipStream_t stream) {
  (void)in_sizes; (void)n_in; (void)out_size; (void)ws_size;
  const float* hidden = (const float*)d_in[0];
  const float* mask   = (const float*)d_in[1];
  const float* q_w = (const float*)d_in[2];
  const float* q_b = (const float*)d_in[3];
  const float* k_w = (const float*)d_in[4];
  const float* v_w = (const float*)d_in[5];
  const float* v_b = (const float*)d_in[6];
  const float* o_w = (const float*)d_in[7];
  const float* o_b = (const float*)d_in[8];

  char* p = (char*)d_ws;
  size_t off = 0;
  auto take = [&](size_t bytes) {
    char* r = p + off;
    off += (bytes + 255) & ~(size_t)255;
    return r;
  };
  bf16* Xbf  = (bf16*)take((size_t)kM * kE * 2);
  bf16* qwbf = (bf16*)take((size_t)kE * kE * 2);
  bf16* kwbf = (bf16*)take((size_t)kE * kE * 2);
  bf16* vwbf = (bf16*)take((size_t)kE * kE * 2);
  bf16* owbf = (bf16*)take((size_t)kE * kE * 2);
  bf16* Qb   = (bf16*)take((size_t)kB * kH * kS * kD * 2);
  bf16* Kbuf = (bf16*)take((size_t)kB * kH * kS * kD * 2);
  bf16* Vt   = (bf16*)take((size_t)kB * kH * kD * kSP * 2);
  bf16* Abf  = (bf16*)take((size_t)kM * kE * 2);

  const int nX8 = (kM * kE) / 8;
  const int nW8 = (kE * kE) / 8;
  cvt_f32_to_bf16_x8<<<(nX8 + 255) / 256, 256, 0, stream>>>(hidden, Xbf, nX8);
  cvt_f32_to_bf16_x8<<<(nW8 + 255) / 256, 256, 0, stream>>>(q_w, qwbf, nW8);
  cvt_f32_to_bf16_x8<<<(nW8 + 255) / 256, 256, 0, stream>>>(k_w, kwbf, nW8);
  cvt_f32_to_bf16_x8<<<(nW8 + 255) / 256, 256, 0, stream>>>(v_w, vwbf, nW8);
  cvt_f32_to_bf16_x8<<<(nW8 + 255) / 256, 256, 0, stream>>>(o_w, owbf, nW8);

  // zero V^T (covers the SP padding so p==0 * pad stays finite)
  hipMemsetAsync(Vt, 0, (size_t)kB * kH * kD * kSP * 2, stream);

  const int gemm_blocks = (kMT * 16 + 7) / 8;  // 376 blocks of 8 waves
  gemm_bf16_32x64<<<gemm_blocks, 256, 0, stream>>>(Xbf, qwbf, q_b, Qb, kScale, 0);
  gemm_bf16_32x64<<<gemm_blocks, 256, 0, stream>>>(Xbf, kwbf, nullptr, Kbuf, 1.0f, 1);
  gemm_bf16_32x64<<<gemm_blocks, 256, 0, stream>>>(Xbf, vwbf, v_b, Vt, 1.0f, 2);

  flash_attn<<<kB * kH * 12, 256, 0, stream>>>(Qb, Kbuf, Vt, mask, Abf);

  gemm_bf16_32x64<<<gemm_blocks, 256, 0, stream>>>(Abf, owbf, o_b, d_out, 1.0f, 3);
}
